// GroupedQueryAttention_68083821576494
// MI455X (gfx1250) — matrix operand order repaired and checked by reading, not compiled, not run on hardware
//
#include <hip/hip_runtime.h>

typedef __bf16 bf16_t;
typedef __attribute__((ext_vector_type(16))) __bf16 v16bf;
typedef __attribute__((ext_vector_type(8)))  float  v8f;
typedef __attribute__((ext_vector_type(4)))  unsigned int u32x4;
typedef __attribute__((ext_vector_type(4)))  int i32x4;

#define HQ  16   // query heads
#define KVH 4    // kv heads
#define HD  64   // head dim

#define AS1 __attribute__((address_space(1)))
#define AS3 __attribute__((address_space(3)))

#if __has_builtin(__builtin_amdgcn_global_load_async_to_lds_b128) && \
    __has_builtin(__builtin_amdgcn_s_wait_asynccnt)
#define USE_ASYNC_LDS 1
#else
#define USE_ASYNC_LDS 0
#endif

// ---------------------------------------------------------------------------
// WMMA fragment loaders (CDNA5 wave32 layouts, CDNA5 ISA 7.12.2)
// A 16x32 bf16: lane L holds row m=L&15; VGPR v holds K pair:
//   k0 = (v/4)*16 + (L>>4)*8 + (v%4)*2
// B 32x16 bf16: lane L holds col n=L&15; VGPR v holds K pair at same k0 as A
// Pairs are contiguous in K => b128-coalescible loads when source is K-major.
// ---------------------------------------------------------------------------
static __device__ inline v16bf load_a_frag(const bf16_t* mat, int ld, int m,
                                           int kbase, int lane) {
  union { v16bf v; unsigned u[8]; } f;
  const bf16_t* row = mat + (long)m * ld;
  int hf = lane >> 4;
#pragma unroll
  for (int vi = 0; vi < 8; ++vi) {
    int k = kbase + ((vi >> 2) << 4) + (hf << 3) + ((vi & 3) << 1);
    f.u[vi] = *(const unsigned*)(row + k);
  }
  return f.v;
}

static __device__ inline v16bf load_b_frag(const bf16_t* mat, int ld, int n,
                                           int kbase, int lane) {
  union { v16bf v; unsigned u[8]; } f;
  const bf16_t* row = mat + (long)n * ld;
  int kb = kbase + ((lane >> 4) << 3);
#pragma unroll
  for (int vi = 0; vi < 8; ++vi)
    f.u[vi] = *(const unsigned*)(row + kb + ((vi >> 2) << 4) + ((vi & 3) << 1));
  return f.v;
}

// ---------------------------------------------------------------------------
// fp32 -> bf16 convert (grid-stride)
// ---------------------------------------------------------------------------
__global__ void cvt_kernel(const float* __restrict__ in, bf16_t* __restrict__ out,
                           long n) {
  long stride = (long)gridDim.x * blockDim.x;
  for (long i = (long)blockIdx.x * blockDim.x + threadIdx.x; i < n; i += stride)
    out[i] = (bf16_t)in[i];
}

// ---------------------------------------------------------------------------
// V: [(b*N+n)*KV + kv, d] f32  ->  [b, kv, d, n] bf16 (key-contiguous)
// ---------------------------------------------------------------------------
__global__ void vtrans_kernel(const float* __restrict__ v, bf16_t* __restrict__ o,
                              int B, int N) {
  long total = (long)B * N * KVH * HD;
  long stride = (long)gridDim.x * blockDim.x;
  for (long i = (long)blockIdx.x * blockDim.x + threadIdx.x; i < total; i += stride) {
    int d = (int)(i & (HD - 1));
    long t = i >> 6;
    int kv = (int)(t % KVH); t /= KVH;
    int n  = (int)(t % N);
    int b  = (int)(t / N);
    o[(((long)b * KVH + kv) * HD + d) * N + n] = (bf16_t)v[i];
  }
}

// ---------------------------------------------------------------------------
// bf16 GEMM with async-LDS double buffering:
//   C[M,Nout] (f32) = A[M,K] * W[Nout,K]^T + bias
// block = 256 threads = 8 waves; block tile = 32(M) x 256(N); K-step = 64.
// Per step: A tile 4KB + W tile 32KB = 72 x 512B segments; each wave issues
// exactly 9 global_load_async_to_lds_b128 => s_wait_asynccnt(9) handshake.
// ---------------------------------------------------------------------------
#define TKS 64

__global__ void gemm_bf16(const bf16_t* __restrict__ A, const bf16_t* __restrict__ W,
                          const float* __restrict__ bias, float* __restrict__ C,
                          int M, int Nout, int K) {
  __shared__ __align__(16) bf16_t lA[2][32 * TKS];    // 2 x 4KB
  __shared__ __align__(16) bf16_t lW[2][256 * TKS];   // 2 x 32KB
  int lane = threadIdx.x & 31;
  int wave = threadIdx.x >> 5;
  int m0 = blockIdx.y * 32;
  int n0 = blockIdx.x * 256;
  int wm = (wave >> 2) * 16;
  int wn = (wave & 3) * 64;

  const char* Ab = (const char*)A;
  const char* Wb = (const char*)W;

  auto stage = [&](int kb, int buf) {
#pragma unroll
    for (int i = 0; i < 9; ++i) {
      int seg = wave + i * 8;            // wave-uniform
      const char* src;
      char* dst;
      if (seg < 8) {                      // A tile: 8 segs x 512B
        int flat = seg * 512 + lane * 16;
        int arow = flat >> 7;             // 128B per row (64 bf16)
        int koff = flat & 127;
        src = Ab + (((long)(m0 + arow) * K + kb) << 1) + koff;
        dst = (char*)&lA[buf][0] + flat;
      } else {                            // W tile: 64 segs x 512B
        int flat = (seg - 8) * 512 + lane * 16;
        int wrow = flat >> 7;
        int koff = flat & 127;
        src = Wb + (((long)(n0 + wrow) * K + kb) << 1) + koff;
        dst = (char*)&lW[buf][0] + flat;
      }
#if USE_ASYNC_LDS
      __builtin_amdgcn_global_load_async_to_lds_b128(
          (AS1 i32x4*)(void*)const_cast<char*>(src),
          (AS3 i32x4*)(void*)dst, 0, 0);
#else
      *(u32x4*)dst = *(const u32x4*)src;
#endif
    }
  };

  v8f acc[4] = {};
  int nsteps = K / TKS;

  stage(0, 0);
  for (int s = 0; s < nsteps; ++s) {
    int buf = s & 1;
    if (s + 1 < nsteps) {
      stage((s + 1) * TKS, buf ^ 1);
#if USE_ASYNC_LDS
      __builtin_amdgcn_s_wait_asynccnt(9);   // previous stage's 9 are done
#endif
    } else {
#if USE_ASYNC_LDS
      __builtin_amdgcn_s_wait_asynccnt(0);
#endif
    }
    __syncthreads();
    const bf16_t* Abuf = &lA[buf][0];
    const bf16_t* Wbuf = &lW[buf][0];
#pragma unroll
    for (int kk = 0; kk < TKS; kk += 32) {
      v16bf a = load_a_frag(Abuf, TKS, wm + (lane & 15), kk, lane);
#pragma unroll
      for (int j = 0; j < 4; ++j) {
        v16bf b = load_b_frag(Wbuf, TKS, wn + j * 16 + (lane & 15), kk, lane);
        acc[j] = __builtin_amdgcn_wmma_f32_16x16x32_bf16(
            false, a, false, b, (short)0, acc[j], false, false);
      }
    }
    __syncthreads();   // protect buffer reuse (staged again at s+2)
  }

  int hf = lane >> 4;
#pragma unroll
  for (int j = 0; j < 4; ++j) {
    int col = n0 + wn + j * 16 + (lane & 15);
    float bv = bias[col];
#pragma unroll
    for (int r = 0; r < 8; ++r) {
      int row = m0 + wm + r + hf * 8;
      C[(long)row * Nout + col] = acc[j][r] + bv;
    }
  }
}

// ---------------------------------------------------------------------------
// Fused RMSNorm + RoPE: in f32 [rows, HD], out bf16 [b, head, n, HD]
// rows flat-index: rid = (b*N + n)*heads + h ; 4 rows of 64 per block
// ---------------------------------------------------------------------------
__global__ void norm_rope_kernel(const float* __restrict__ inp,
                                 const float* __restrict__ nw,
                                 const float* __restrict__ sint,
                                 const float* __restrict__ cost,
                                 bf16_t* __restrict__ outp, int heads, int N) {
  __shared__ float red[4][HD];
  __shared__ float nb[4][HD];
  int t = threadIdx.x;
  int sub = t >> 6, d = t & 63;
  long rid = (long)blockIdx.x * 4 + sub;
  int h = (int)(rid % heads);
  long t2 = rid / heads;
  int n = (int)(t2 % N);
  int b = (int)(t2 / N);
  float val = inp[rid * HD + d];
  red[sub][d] = val * val;
  __syncthreads();
#pragma unroll
  for (int s = 32; s > 0; s >>= 1) {
    if (d < s) red[sub][d] += red[sub][d + s];
    __syncthreads();
  }
  float rms = rsqrtf(red[sub][0] * (1.0f / HD) + 1e-6f);
  float xn = val * rms * nw[d];
  nb[sub][d] = xn;
  __syncthreads();
  float partner = (d < 32) ? -nb[sub][d + 32] : nb[sub][d - 32];
  float o = xn * cost[(long)n * HD + d] + partner * sint[(long)n * HD + d];
  outp[(((long)b * heads + h) * N + n) * HD + d] = (bf16_t)o;
}

// ---------------------------------------------------------------------------
// Flash attention: one wave per (b, h, 16-query tile); streams 32 keys/step.
// Q bf16 [b,h,n,d]; K bf16 [b,kv,n,d]; Vt bf16 [b,kv,d,n]; O bf16 [b,n, h*d]
// ---------------------------------------------------------------------------
__global__ void attn_kernel(const bf16_t* __restrict__ Qm,
                            const bf16_t* __restrict__ Km,
                            const bf16_t* __restrict__ Vt,
                            bf16_t* __restrict__ Om, int B, int N) {
  __shared__ bf16_t plds[8][16 * 32];   // per-wave P staging (layout transpose)
  int lane = threadIdx.x & 31;
  int wave = threadIdx.x >> 5;
  int ntiles = N >> 4;
  long tile = (long)blockIdx.x * 8 + wave;
  int qt = (int)(tile % ntiles);
  int h  = (int)((tile / ntiles) % HQ);
  int b  = (int)(tile / ((long)ntiles * HQ));
  int kvh = h / (HQ / KVH);
  const bf16_t* Qh = Qm + (((long)b * HQ + h) * N + (long)qt * 16) * HD;
  const bf16_t* Kh = Km + ((long)b * KVH + kvh) * N * HD;
  const bf16_t* Vh = Vt + ((long)b * KVH + kvh) * HD * N;
  bf16_t* pl = &plds[wave][0];

  v16bf qa0 = load_a_frag(Qh, HD, lane & 15, 0, lane);
  v16bf qa1 = load_a_frag(Qh, HD, lane & 15, 32, lane);

  v8f out0 = {}, out1 = {}, out2 = {}, out3 = {};
  float mrow[8], lrow[8];
#pragma unroll
  for (int r = 0; r < 8; ++r) { mrow[r] = -3.0e38f; lrow[r] = 0.0f; }

  const float scale = 0.125f;  // 1/sqrt(64)
  int hf = lane >> 4;

  for (int kc = 0; kc < N; kc += 32) {
    if (kc + 32 < N) {   // prefetch next key/value chunk into cache
      __builtin_prefetch((const void*)(Kh + (long)(kc + 32) * HD), 0, 1);
      __builtin_prefetch((const void*)(Vh + (long)(kc + 32)), 0, 1);
    }
    // ---- S = Q * K^T for two 16-key sub-tiles (D=64 => 2 k-steps each) ----
    v8f s0 = {}, s1 = {};
    {
      v16bf kb = load_b_frag(Kh, HD, kc + (lane & 15), 0, lane);
      s0 = __builtin_amdgcn_wmma_f32_16x16x32_bf16(false, qa0, false, kb, (short)0, s0, false, false);
      kb = load_b_frag(Kh, HD, kc + (lane & 15), 32, lane);
      s0 = __builtin_amdgcn_wmma_f32_16x16x32_bf16(false, qa1, false, kb, (short)0, s0, false, false);
      kb = load_b_frag(Kh, HD, kc + 16 + (lane & 15), 0, lane);
      s1 = __builtin_amdgcn_wmma_f32_16x16x32_bf16(false, qa0, false, kb, (short)0, s1, false, false);
      kb = load_b_frag(Kh, HD, kc + 16 + (lane & 15), 32, lane);
      s1 = __builtin_amdgcn_wmma_f32_16x16x32_bf16(false, qa1, false, kb, (short)0, s1, false, false);
    }
    // ---- online softmax (row = r + 8*(lane>>4); cols across a 16-lane half)
#pragma unroll
    for (int r = 0; r < 8; ++r) {
      float a = s0[r] * scale;
      float c = s1[r] * scale;
      float mx = fmaxf(a, c);
      mx = fmaxf(mx, __shfl_xor(mx, 1, 32));
      mx = fmaxf(mx, __shfl_xor(mx, 2, 32));
      mx = fmaxf(mx, __shfl_xor(mx, 4, 32));
      mx = fmaxf(mx, __shfl_xor(mx, 8, 32));
      float mnew = fmaxf(mrow[r], mx);
      float sc = __expf(mrow[r] - mnew);
      float p0 = __expf(a - mnew);
      float p1 = __expf(c - mnew);
      float rs = p0 + p1;
      rs += __shfl_xor(rs, 1, 32);
      rs += __shfl_xor(rs, 2, 32);
      rs += __shfl_xor(rs, 4, 32);
      rs += __shfl_xor(rs, 8, 32);
      lrow[r] = lrow[r] * sc + rs;
      mrow[r] = mnew;
      out0[r] *= sc; out1[r] *= sc; out2[r] *= sc; out3[r] *= sc;
      pl[(r + hf * 8) * 32 + (lane & 15)]      = (bf16_t)p0;
      pl[(r + hf * 8) * 32 + 16 + (lane & 15)] = (bf16_t)p1;
    }
    // ---- O += P * V  (P 16x32 from LDS, V 32xD via transposed layout) ----
    v16bf pa = load_a_frag(pl, 32, lane & 15, 0, lane);
    v16bf vb = load_b_frag(Vh, N, 0 * 16 + (lane & 15), kc, lane);
    out0 = __builtin_amdgcn_wmma_f32_16x16x32_bf16(false, pa, false, vb, (short)0, out0, false, false);
    vb = load_b_frag(Vh, N, 1 * 16 + (lane & 15), kc, lane);
    out1 = __builtin_amdgcn_wmma_f32_16x16x32_bf16(false, pa, false, vb, (short)0, out1, false, false);
    vb = load_b_frag(Vh, N, 2 * 16 + (lane & 15), kc, lane);
    out2 = __builtin_amdgcn_wmma_f32_16x16x32_bf16(false, pa, false, vb, (short)0, out2, false, false);
    vb = load_b_frag(Vh, N, 3 * 16 + (lane & 15), kc, lane);
    out3 = __builtin_amdgcn_wmma_f32_16x16x32_bf16(false, pa, false, vb, (short)0, out3, false, false);
  }
  // ---- finalize: divide by row sum, store bf16 [b, n, h*64 + d] ----
#pragma unroll
  for (int r = 0; r < 8; ++r) {
    long row = (long)b * N + (long)qt * 16 + r + hf * 8;
    float inv = 1.0f / lrow[r];
    long base = row * (HQ * HD) + (long)h * HD + (lane & 15);
    Om[base +  0] = (bf16_t)(out0[r] * inv);
    Om[base + 16] = (bf16_t)(out1[r] * inv);
    Om[base + 32] = (bf16_t)(out2[r] * inv);
    Om[base + 48] = (bf16_t)(out3[r] * inv);
  }
}

// ---------------------------------------------------------------------------
extern "C" void kernel_launch(void* const* d_in, const int* in_sizes, int n_in,
                              void* d_out, int out_size, void* d_ws, size_t ws_size,
                              hipStream_t stream) {
  const float* x      = (const float*)d_in[0];
  const float* sint   = (const float*)d_in[1];
  const float* cost   = (const float*)d_in[2];
  const float* wq_w   = (const float*)d_in[3];
  const float* wq_b   = (const float*)d_in[4];
  const float* wk_w   = (const float*)d_in[5];
  const float* wk_b   = (const float*)d_in[6];
  const float* wv_w   = (const float*)d_in[7];
  const float* wv_b   = (const float*)d_in[8];
  const float* qn_w   = (const float*)d_in[9];
  const float* kn_w   = (const float*)d_in[10];
  const float* proj_w = (const float*)d_in[11];
  const float* proj_b = (const float*)d_in[12];

  const int E   = in_sizes[4];                               // 1024
  const int KVD = in_sizes[6];                               // 256
  const int N   = in_sizes[1] / HD;                          // 2048
  const int B   = (int)((long)in_sizes[0] / ((long)N * E));  // 2
  const long M  = (long)B * N;                               // 4096

  char* ws = (char*)d_ws;
  size_t off = 0;
  auto alloc = [&](size_t bytes) {
    char* p = ws + off;
    off = (off + bytes + 255) & ~(size_t)255;
    return p;
  };
  bf16_t* xb  = (bf16_t*)alloc((size_t)M * E * 2);
  bf16_t* wqb = (bf16_t*)alloc((size_t)E * E * 2);
  bf16_t* wkb = (bf16_t*)alloc((size_t)KVD * E * 2);
  bf16_t* wvb = (bf16_t*)alloc((size_t)KVD * E * 2);
  bf16_t* prb = (bf16_t*)alloc((size_t)E * E * 2);
  float*  qf  = (float*)alloc((size_t)M * E * 4);
  float*  kf  = (float*)alloc((size_t)M * KVD * 4);
  float*  vf  = (float*)alloc((size_t)M * KVD * 4);
  bf16_t* qb  = (bf16_t*)alloc((size_t)M * E * 2);
  bf16_t* kb  = (bf16_t*)alloc((size_t)M * KVD * 2);
  bf16_t* vtb = (bf16_t*)alloc((size_t)M * KVD * 2);
  bf16_t* aob = (bf16_t*)alloc((size_t)M * E * 2);
  (void)n_in; (void)out_size; (void)ws_size;

  // 1) bf16 conversions
  cvt_kernel<<<1024, 256, 0, stream>>>(x, xb, M * E);
  cvt_kernel<<<512, 256, 0, stream>>>(wq_w, wqb, (long)E * E);
  cvt_kernel<<<256, 256, 0, stream>>>(wk_w, wkb, (long)KVD * E);
  cvt_kernel<<<256, 256, 0, stream>>>(wv_w, wvb, (long)KVD * E);
  cvt_kernel<<<512, 256, 0, stream>>>(proj_w, prb, (long)E * E);

  // 2) QKV projections (WMMA bf16, async-LDS double buffered)
  gemm_bf16<<<dim3(E / 256, (int)(M / 32)), 256, 0, stream>>>(xb, wqb, wq_b, qf, (int)M, E, E);
  gemm_bf16<<<dim3(KVD / 256, (int)(M / 32)), 256, 0, stream>>>(xb, wkb, wk_b, kf, (int)M, KVD, E);
  gemm_bf16<<<dim3(KVD / 256, (int)(M / 32)), 256, 0, stream>>>(xb, wvb, wv_b, vf, (int)M, KVD, E);

  // 3) norm + rope + layout conversion
  norm_rope_kernel<<<(int)(M * HQ / 4), 256, 0, stream>>>(qf, qn_w, sint, cost, qb, HQ, N);
  norm_rope_kernel<<<(int)(M * KVH / 4), 256, 0, stream>>>(kf, kn_w, sint, cost, kb, KVH, N);
  vtrans_kernel<<<1024, 256, 0, stream>>>(vf, vtb, B, N);

  // 4) flash attention (WMMA bf16)
  attn_kernel<<<(int)((long)B * HQ * (N / 16) / 8), 256, 0, stream>>>(qb, kb, vtb, aob, B, N);

  // 5) output projection (WMMA bf16) -> f32 d_out
  gemm_bf16<<<dim3(E / 256, (int)(M / 32)), 256, 0, stream>>>(aob, prb, proj_b, (float*)d_out, (int)M, E, E);
}
